// KeyValueMemory_34797825032309
// MI455X (gfx1250) — matrix operand order repaired and checked by reading, not compiled, not run on hardware
//
#include <hip/hip_runtime.h>
#include <hip/hip_bf16.h>

// CDNA5 / gfx1250, wave32 WMMA types
typedef __bf16 v16bf __attribute__((ext_vector_type(16)));
typedef __bf16 v8bf  __attribute__((ext_vector_type(8)));
typedef float  v8f   __attribute__((ext_vector_type(8)));

#define KDIM   128
#define VDIM   1024
#define NSLOTS 512
#define TOPK   8
#define MT     16   // token rows per block

#define WQ_ELEMS   (KDIM * VDIM)    // 131072
#define KEYS_ELEMS (NSLOTS * KDIM)  //  65536
#define VALT_ELEMS (VDIM * NSLOTS)  // 524288

// Build a 16-element bf16 fragment from two contiguous 16B chunks (global/L2).
__device__ __forceinline__ v16bf frag_contig(const __bf16* p) {
    const v8bf* q = (const v8bf*)p;
    v8bf lo = q[0], hi = q[1];
    return __builtin_shufflevector(lo, hi, 0,1,2,3,4,5,6,7,8,9,10,11,12,13,14,15);
}
// A-fragment: lane holds K run [kbA, kbA+16) -> chunks 0 and 1.
__device__ __forceinline__ v16bf fragA_lds(const __bf16* p) {
    const v8bf* q = (const v8bf*)p;
    v8bf lo = q[0], hi = q[1];
    return __builtin_shufflevector(lo, hi, 0,1,2,3,4,5,6,7,8,9,10,11,12,13,14,15);
}

// ---------------------------------------------------------------------------
// Kernel 0: one-shot bf16 conversion of the small L2-resident parameter
// matrices into workspace. valT is values transposed: valT[n][k].
// ---------------------------------------------------------------------------
__global__ __launch_bounds__(256) void kv_convert(
    const float* __restrict__ Wq, const float* __restrict__ keys,
    const float* __restrict__ values,
    __bf16* __restrict__ wqbf, __bf16* __restrict__ keysbf,
    __bf16* __restrict__ valT)
{
    const int i      = blockIdx.x * 256 + threadIdx.x;
    const int stride = gridDim.x * 256;
    for (int idx = i; idx < WQ_ELEMS; idx += stride)
        wqbf[idx] = (__bf16)Wq[idx];                      // [k][v], row-major
    for (int idx = i; idx < KEYS_ELEMS; idx += stride)
        keysbf[idx] = (__bf16)keys[idx];                  // [n][k], row-major
    for (int idx = i; idx < VALT_ELEMS; idx += stride) {
        int n = idx >> 9, k = idx & (NSLOTS - 1);
        valT[idx] = (__bf16)values[(size_t)k * VDIM + n]; // transpose
    }
}

// ---------------------------------------------------------------------------
// Kernel 1: fused  q = query*Wq^T + bq  ->  scores = q*keys^T * scale
//           -> top-8 threshold -> softmax -> weights (f32, to d_out tail)
// Block: 256 threads (8 waves), MT=16 token rows.
// Wave w owns q columns [16w,16w+16) and score columns [64w,64w+64).
// B-fragments stream straight from L2-resident bf16 copies (contiguous 32B
// per lane); A-fragments are 2x ds_load_b128 from LDS. Only 3 barriers.
// ---------------------------------------------------------------------------
__global__ __launch_bounds__(256) void kv_scores_softmax(
    const float*  __restrict__ query, const __bf16* __restrict__ wqbf,
    const __bf16* __restrict__ keysbf, const float* __restrict__ bq,
    float* __restrict__ weights_out)
{
    __shared__ __bf16 qbf[MT][VDIM];     // 32 KB: query rows (bf16)
    __shared__ __bf16 qk[MT][KDIM];      //  4 KB: projected q (bf16)
    __shared__ float  sc[MT][NSLOTS];    // 32 KB: scores (f32)

    const int tid  = threadIdx.x;
    const int wave = tid >> 5;
    const int lane = tid & 31;
    const int m0   = blockIdx.x * MT;

    // stage 16 query rows, f32 -> bf16 (streaming HBM read, b128 loads)
    {
        const float4* qv = (const float4*)(query + (size_t)m0 * VDIM);
        #pragma unroll
        for (int t = 0; t < 16; ++t) {
            int fi  = tid + t * 256;         // 0..4095 float4s
            int row = fi >> 8;               // 256 float4 per row
            int c4  = (fi & 255) * 4;
            float4 v = qv[fi];
            qbf[row][c4 + 0] = (__bf16)v.x;
            qbf[row][c4 + 1] = (__bf16)v.y;
            qbf[row][c4 + 2] = (__bf16)v.z;
            qbf[row][c4 + 3] = (__bf16)v.w;
        }
    }

    // fragment geometry (wave32), contiguous K order: k = 16*(lane/16) + s
    const int mA   = lane & 15;                 // A: M = lane%16
    const int kbA  = (lane < 16) ? 0 : 16;      // A: K base per half-wave
    const int kbB  = (lane < 16) ? 0 : 16;      // B: K base per half-wave
    const int nloc = lane & 15;                 // B/C/D: N = lane%16

    __syncthreads();

    // ----- GEMM 1: q tile (16x16 per wave), K=VDIM in 32 steps, no barriers --
    {
        const __bf16* brow = wqbf + (size_t)(wave * 16 + nloc) * VDIM + kbB;
        v8f qacc = {};
        #pragma unroll 4
        for (int vb = 0; vb < VDIM; vb += 32) {
            v16bf a = fragA_lds(&qbf[mA][vb + kbA]);
            v16bf b = frag_contig(brow + vb);
            qacc = __builtin_amdgcn_wmma_f32_16x16x32_bf16(
                false, a, false, b, (short)0, qacc, false, false);
        }
        // write q tile (+bias) as bf16 (C layout: lane=N, vgpr r=M row)
        float bias = bq[wave * 16 + nloc];
        #pragma unroll
        for (int r = 0; r < 8; ++r) {
            int Mrow = r + ((lane < 16) ? 0 : 8);
            qk[Mrow][wave * 16 + nloc] = (__bf16)(qacc[r] + bias);
        }
    }
    __syncthreads();

    // ----- GEMM 2: scores (16x64 per wave), K=128 in 4 steps -----
    {
        const float scale = 0.08838834764831845f;   // 1/sqrt(128)
        v16bf aS[4];
        #pragma unroll
        for (int kt = 0; kt < 4; ++kt)
            aS[kt] = fragA_lds(&qk[mA][kt * 32 + kbA]);

        #pragma unroll
        for (int nt = 0; nt < 4; ++nt) {
            int n = wave * 64 + nt * 16 + nloc;     // slot this lane produces
            const __bf16* krow = keysbf + (size_t)n * KDIM + kbB; // L2-resident
            v8f sacc = {};
            #pragma unroll
            for (int kt = 0; kt < 4; ++kt) {
                v16bf b = frag_contig(krow + kt * 32);
                sacc = __builtin_amdgcn_wmma_f32_16x16x32_bf16(
                    false, aS[kt], false, b, (short)0, sacc, false, false);
            }
            #pragma unroll
            for (int r = 0; r < 8; ++r) {
                int Mrow = r + ((lane < 16) ? 0 : 8);
                sc[Mrow][n] = sacc[r] * scale;
            }
        }
    }
    __syncthreads();

    // ----- top-8 threshold + softmax; each wave handles 2 rows -----
    for (int rr = 0; rr < 2; ++rr) {
        int row = wave * 2 + rr;
        float loc[16];
        #pragma unroll
        for (int j = 0; j < 16; ++j) loc[j] = sc[row][lane + 32 * j];

        float rowmax = 0.f, thresh = 0.f;
        for (int t = 0; t < TOPK; ++t) {
            float lm = -__builtin_inff(); int li = 0;
            #pragma unroll
            for (int j = 0; j < 16; ++j)
                if (loc[j] > lm) { lm = loc[j]; li = j; }
            float bm = lm; int bl = lane; int bi = li;
            #pragma unroll
            for (int off = 16; off > 0; off >>= 1) {
                float om = __shfl_xor(bm, off, 32);
                int   ol = __shfl_xor(bl, off, 32);
                int   oi = __shfl_xor(bi, off, 32);
                if (om > bm || (om == bm && ol < bl)) { bm = om; bl = ol; bi = oi; }
            }
            if (t == 0) rowmax = bm;
            thresh = bm;                        // after loop: 8th largest
            if (lane == bl) loc[bi] = -__builtin_inff();
        }

        float sum = 0.f;
        float w[16];
        #pragma unroll
        for (int j = 0; j < 16; ++j) {
            float s = sc[row][lane + 32 * j];
            float e = (s >= thresh) ? __expf(s - rowmax) : 0.f;
            w[j] = e; sum += e;
        }
        #pragma unroll
        for (int off = 16; off > 0; off >>= 1) sum += __shfl_xor(sum, off, 32);
        float inv = 1.f / sum;

        float* wrow = weights_out + (size_t)(m0 + row) * NSLOTS;
        #pragma unroll
        for (int j = 0; j < 16; ++j) wrow[lane + 32 * j] = w[j] * inv;
    }
}

// ---------------------------------------------------------------------------
// Kernel 2: retrieved = weights (Mx512) * values (512x1024)
// Grid: (M/16, 8). Wave w -> cols [16w,16w+16) of this block's 128-col strip.
// A (weights) staged once to LDS as bf16; B streams from transposed bf16
// values in L2 (contiguous 32B per lane). K=512 in 16 steps, 1 barrier.
// ---------------------------------------------------------------------------
__global__ __launch_bounds__(256) void kv_retrieve(
    const float* __restrict__ weights, const __bf16* __restrict__ valT,
    float* __restrict__ out)
{
    __shared__ __bf16 wbf[MT][NSLOTS];   // 16 KB: weight rows (bf16)

    const int tid   = threadIdx.x;
    const int wave  = tid >> 5;
    const int lane  = tid & 31;
    const int m0    = blockIdx.x * MT;
    const int nbase = blockIdx.y * 128;

    // stage 16 weight rows, f32 -> bf16
    {
        const float4* wv = (const float4*)(weights + (size_t)m0 * NSLOTS);
        #pragma unroll
        for (int t = 0; t < 8; ++t) {
            int fi  = tid + t * 256;        // 0..2047 float4s
            int row = fi >> 7;              // 128 float4 per row
            int c4  = (fi & 127) * 4;
            float4 v = wv[fi];
            wbf[row][c4 + 0] = (__bf16)v.x;
            wbf[row][c4 + 1] = (__bf16)v.y;
            wbf[row][c4 + 2] = (__bf16)v.z;
            wbf[row][c4 + 3] = (__bf16)v.w;
        }
    }

    const int mA   = lane & 15;
    const int kbA  = (lane < 16) ? 0 : 16;
    const int kbB  = (lane < 16) ? 0 : 16;
    const int nloc = lane & 15;

    const int ncol = nbase + wave * 16 + nloc;
    const __bf16* vrow = valT + (size_t)ncol * NSLOTS + kbB;  // L2-resident

    __syncthreads();

    v8f acc = {};
    #pragma unroll 4
    for (int kb = 0; kb < NSLOTS; kb += 32) {
        if (kb + 32 < NSLOTS)
            __builtin_prefetch(vrow + kb + 32, 0, 0);   // global_prefetch_b8
        v16bf a = fragA_lds(&wbf[mA][kb + kbA]);
        v16bf b = frag_contig(vrow + kb);
        acc = __builtin_amdgcn_wmma_f32_16x16x32_bf16(
            false, a, false, b, (short)0, acc, false, false);
    }

    #pragma unroll
    for (int r = 0; r < 8; ++r) {
        int Mrow = r + ((lane < 16) ? 0 : 8);
        out[(size_t)(m0 + Mrow) * VDIM + nbase + wave * 16 + nloc] = acc[r];
    }
}

// ---------------------------------------------------------------------------
extern "C" void kernel_launch(void* const* d_in, const int* in_sizes, int n_in,
                              void* d_out, int out_size, void* d_ws, size_t ws_size,
                              hipStream_t stream) {
    const float* query  = (const float*)d_in[0];
    const float* keys   = (const float*)d_in[1];
    const float* values = (const float*)d_in[2];
    const float* Wq     = (const float*)d_in[3];
    const float* bq     = (const float*)d_in[4];
    // d_in[5] = top_k scalar (8) — hardwired as TOPK.

    const int M = in_sizes[0] / VDIM;           // 32768 token rows

    float* retrieved = (float*)d_out;                        // M * 1024
    float* weights   = retrieved + (size_t)M * VDIM;         // M * 512

    __bf16* wqbf   = (__bf16*)d_ws;                          // 131072 elems
    __bf16* keysbf = wqbf + WQ_ELEMS;                        //  65536 elems
    __bf16* valT   = keysbf + KEYS_ELEMS;                    // 524288 elems

    kv_convert<<<512, 256, 0, stream>>>(Wq, keys, values, wqbf, keysbf, valT);
    kv_scores_softmax<<<M / MT, 256, 0, stream>>>(query, wqbf, keysbf, bq, weights);
    kv_retrieve<<<dim3(M / MT, VDIM / 128), 256, 0, stream>>>(weights, valT, retrieved);
}
